// MoEGPT_49486613184900
// MI455X (gfx1250) — matrix operand order repaired and checked by reading, not compiled, not run on hardware
//
#include <hip/hip_runtime.h>

typedef __attribute__((ext_vector_type(16))) __bf16 v16bf;
typedef __attribute__((ext_vector_type(8)))  float  v8f;
typedef __attribute__((ext_vector_type(4)))  unsigned int v4u;
typedef __attribute__((ext_vector_type(4)))  unsigned int u32x4;
typedef __attribute__((ext_vector_type(8)))  int i32x8;
typedef __attribute__((ext_vector_type(4)))  int i32x4;

union FragU {
  v16bf v;
  v4u u[2];
  unsigned short s[16];
};

__device__ __forceinline__ unsigned short f2bf(float x) {
  unsigned int u = __float_as_uint(x);
  u += 0x7FFFu + ((u >> 16) & 1u);
  return (unsigned short)(u >> 16);
}
__device__ __forceinline__ float bf2f(unsigned short s) {
  return __uint_as_float(((unsigned int)s) << 16);
}

// A fragment: 16x32 bf16 tile from row-major [*, ld]; lane L holds row (L&15),
// K = kb..kb+7 and kb+16..kb+23 with kb = k + ((L>=16)?8:0)  (ISA 16-bit A layout)
__device__ __forceinline__ v16bf load_fragA(const unsigned short* base, int ld, int row, int k) {
  int lane = threadIdx.x & 31;
  const unsigned short* p = base + (size_t)(row + (lane & 15)) * ld + k + ((lane >> 4) << 3);
  FragU f;
  f.u[0] = *(const v4u*)p;
  f.u[1] = *(const v4u*)(p + 16);
  return f.v;
}

// B fragment: 32x16 bf16; column n = rowN + (L&15) of row-major weight [N, ld];
// K = kb..kb+7 and kb+16..kb+23 with kb = k + ((L>=16)?8:0)  (matches A layout)
__device__ __forceinline__ v16bf load_fragB(const unsigned short* base, int ld, int rowN, int k) {
  int lane = threadIdx.x & 31;
  const unsigned short* p = base + (size_t)(rowN + (lane & 15)) * ld + k + ((lane >> 4) << 3);
  FragU f;
  f.u[0] = *(const v4u*)p;
  f.u[1] = *(const v4u*)(p + 16);
  return f.v;
}

__device__ __forceinline__ v8f wmma_bf16(v16bf a, v16bf b, v8f c) {
  return __builtin_amdgcn_wmma_f32_16x16x32_bf16(false, a, false, b, (short)0, c, false, false);
}

// ---------------- Tensor Data Mover: 2D bf16 tile (rows x cols) global -> LDS ----------------
// D# per CDNA5 ISA: group0 {count, lds_addr, global_addr, type=2},
// group1 {data_size=1(2B), tensor_dim0=cols, tensor_dim1=rows, tile_dim0=cols, tile_dim1=rows,
//         tensor_dim0_stride=pitch}; groups 2/3 zero (2D tile).
__device__ __forceinline__ void tdm_load_tile(unsigned int ldsAddr, const void* gptr,
                                              unsigned int rows, unsigned int cols,
                                              unsigned int pitch) {
#if __has_builtin(__builtin_amdgcn_tensor_load_to_lds)
  unsigned long long ga = (unsigned long long)(size_t)gptr;
  u32x4 g0;
  g0[0] = 1u;                                             // count=1 (valid), user mode
  g0[1] = ldsAddr;                                        // LDS byte address
  g0[2] = (unsigned int)(ga & 0xFFFFFFFFu);               // global_addr[31:0]
  g0[3] = (unsigned int)((ga >> 32) & 0x1FFFFFFu) | (2u << 30);  // global_addr[56:32] | type=2
  i32x8 g1;
  g1[0] = (int)(1u << 16);                                // workgroup_mask=0, data_size=1 (2B)
  g1[1] = (int)((cols & 0xFFFFu) << 16);                  // tensor_dim0[15:0]
  g1[2] = (int)(((cols >> 16) & 0xFFFFu) | ((rows & 0xFFFFu) << 16)); // dim0 hi | dim1 lo
  g1[3] = (int)(((rows >> 16) & 0xFFFFu) | ((cols & 0xFFFFu) << 16)); // dim1 hi | tile_dim0
  g1[4] = (int)(rows & 0xFFFFu);                          // tile_dim1 | tile_dim2=0
  g1[5] = (int)pitch;                                     // tensor_dim0_stride[31:0]
  g1[6] = 0;
  g1[7] = 0;
  i32x4 z4 = {0, 0, 0, 0};
  i32x8 z8 = {0, 0, 0, 0, 0, 0, 0, 0};
  __builtin_amdgcn_tensor_load_to_lds(g0, g1, z4, z4, z8, 0);
#endif
}

// ---------------- elementwise / small kernels ----------------

__global__ void cvt_kernel(const float* __restrict__ in, unsigned short* __restrict__ out, long n) {
  long i = (long)blockIdx.x * blockDim.x + threadIdx.x;
  long stride = (long)gridDim.x * blockDim.x;
  for (; i < n; i += stride) out[i] = f2bf(in[i]);
}

__global__ void embed_kernel(const int* __restrict__ idx, const float* __restrict__ wte,
                             const float* __restrict__ wpe, float* __restrict__ x, int T, int C) {
  int row = blockIdx.x;           // b*T + t
  int t = row % T;
  int tok = idx[row];
  const float* a = wte + (size_t)tok * C;
  const float* p = wpe + (size_t)t * C;
  float* o = x + (size_t)row * C;
  for (int i = threadIdx.x; i < C; i += blockDim.x) o[i] = a[i] + p[i];
}

// one wave per row of 1024; out bf16 or f32; input row = rowOffset + r*rowStride
__global__ void ln_kernel(const float* __restrict__ x, const float* __restrict__ w, void* out,
                          int nrows, int rowOffset, int rowStride, int outBf16) {
  int lane = threadIdx.x & 31;
  int wid = threadIdx.x >> 5;
  int r = blockIdx.x * (blockDim.x >> 5) + wid;
  if (r >= nrows) return;
  const float* px = x + ((size_t)rowOffset + (size_t)r * rowStride) * 1024;
  float vals[32];
  float s = 0.f, s2 = 0.f;
  for (int i = 0; i < 32; ++i) {
    float v = px[lane + i * 32];
    vals[i] = v; s += v; s2 += v * v;
  }
  for (int m = 16; m >= 1; m >>= 1) { s += __shfl_xor(s, m, 32); s2 += __shfl_xor(s2, m, 32); }
  float mean = s * (1.f / 1024.f);
  float var = s2 * (1.f / 1024.f) - mean * mean;
  float inv = rsqrtf(var + 1e-5f);
  if (outBf16) {
    unsigned short* o = (unsigned short*)out + (size_t)r * 1024;
    for (int i = 0; i < 32; ++i) {
      int c = lane + i * 32;
      o[c] = f2bf((vals[i] - mean) * inv * w[c]);
    }
  } else {
    float* o = (float*)out + (size_t)r * 1024;
    for (int i = 0; i < 32; ++i) {
      int c = lane + i * 32;
      o[c] = (vals[i] - mean) * inv * w[c];
    }
  }
}

// one wave per token: 8 expert logits, softmax, top-2, normalized coefs (0 for unselected)
__global__ void router_kernel(const unsigned short* __restrict__ xln, const float* __restrict__ gate,
                              float* __restrict__ coef, int ntok) {
  int lane = threadIdx.x & 31;
  int wid = threadIdx.x >> 5;
  int t = blockIdx.x * (blockDim.x >> 5) + wid;
  if (t >= ntok) return;
  const unsigned short* px = xln + (size_t)t * 1024;
  float pr[8];
  for (int e = 0; e < 8; ++e) {
    float p = 0.f;
    const float* g = gate + e * 1024;
    for (int k = lane; k < 1024; k += 32) p += bf2f(px[k]) * g[k];
    for (int m = 16; m >= 1; m >>= 1) p += __shfl_xor(p, m, 32);
    pr[e] = p;
  }
  float mx = pr[0];
  for (int e = 1; e < 8; ++e) mx = fmaxf(mx, pr[e]);
  float se = 0.f;
  for (int e = 0; e < 8; ++e) { pr[e] = __expf(pr[e] - mx); se += pr[e]; }
  for (int e = 0; e < 8; ++e) pr[e] /= se;
  int i1 = 0;
  for (int e = 1; e < 8; ++e) if (pr[e] > pr[i1]) i1 = e;
  int i2 = (i1 == 0) ? 1 : 0;
  for (int e = 0; e < 8; ++e) if (e != i1 && pr[e] > pr[i2]) i2 = e;
  float norm = pr[i1] + pr[i2];
  if (lane < 8) {
    float c = (lane == i1) ? pr[i1] / norm : (lane == i2) ? pr[i2] / norm : 0.f;
    coef[(size_t)t * 8 + lane] = c;
  }
}

// ---------------- WMMA GEMM: D[M,N] = A[M,K](bf16) @ W[N,K]^T(bf16) ----------------
// block = 256 threads (8 waves as 2x4), wave tile 64x32, block tile 128x128.
// TDM double-buffered LDS staging: wave 0 issues tensor_load_to_lds for the next
// 128x32 A and B tiles while all waves consume the current tiles via ds loads.
// mode 0: f32 out   1: bf16 out   2: gelu->bf16   3: f32 out + residual R
// mode 4: f32 out = R + coef[row,expert]*acc   (MoE accumulate)
__global__ void __launch_bounds__(256, 1)
gemm_kernel(const unsigned short* __restrict__ A, const unsigned short* __restrict__ W,
            int M, int N, int K, void* D, const float* __restrict__ R,
            const float* __restrict__ coef, int expert, int mode) {
  __shared__ __align__(16) unsigned short lA[2][128 * 32];
  __shared__ __align__(16) unsigned short lB[2][128 * 32];

  int lane = threadIdx.x & 31;
  int wid = threadIdx.x >> 5;
  int bmBlk = blockIdx.y * 128;
  int bnBlk = blockIdx.x * 128;
  int wm = (wid >> 2) * 64;
  int wn = (wid & 3) * 32;

  v8f acc[4][2];
  for (int i = 0; i < 4; ++i)
    for (int j = 0; j < 2; ++j)
      for (int r = 0; r < 8; ++r) acc[i][j][r] = 0.f;

  const unsigned short* gA = A + (size_t)bmBlk * K;
  const unsigned short* gW = W + (size_t)bnBlk * K;
  int nk = K >> 5;

#if __has_builtin(__builtin_amdgcn_tensor_load_to_lds)
  if (wid == 0) {
    tdm_load_tile((unsigned int)(size_t)&lA[0][0], gA, 128, 32, (unsigned int)K);
    tdm_load_tile((unsigned int)(size_t)&lB[0][0], gW, 128, 32, (unsigned int)K);
  }
  for (int ik = 0; ik < nk; ++ik) {
    int cur = ik & 1;
    if (wid == 0) {
      if (ik + 1 < nk) {
        tdm_load_tile((unsigned int)(size_t)&lA[cur ^ 1][0], gA + (size_t)(ik + 1) * 32, 128, 32, (unsigned int)K);
        tdm_load_tile((unsigned int)(size_t)&lB[cur ^ 1][0], gW + (size_t)(ik + 1) * 32, 128, 32, (unsigned int)K);
        __builtin_amdgcn_s_wait_tensorcnt((short)2);   // current A+B complete
      } else {
        __builtin_amdgcn_s_wait_tensorcnt((short)0);
      }
    }
    __syncthreads();   // publish current tiles to all waves
    v16bf af[4], bfr[2];
    for (int i = 0; i < 4; ++i) af[i]  = load_fragA(&lA[cur][0], 32, wm + i * 16, 0);
    for (int j = 0; j < 2; ++j) bfr[j] = load_fragB(&lB[cur][0], 32, wn + j * 16, 0);
    for (int i = 0; i < 4; ++i)
      for (int j = 0; j < 2; ++j)
        acc[i][j] = wmma_bf16(af[i], bfr[j], acc[i][j]);
    __syncthreads();   // all reads done before buffer is overwritten
  }
#else
  // fallback: cooperative copy global -> LDS (2x b128 per thread per tile)
  for (int ik = 0; ik < nk; ++ik) {
    int cur = ik & 1;
    {
      int tt = (int)threadIdx.x;           // 256 threads: 2 chunks of 16B each per tile
      for (int c = 0; c < 2; ++c) {
        int elt = (tt * 2 + c) * 8;        // 8 ushorts per chunk, 4096 elems per tile
        int rw = elt >> 5, cl = elt & 31;
        *(v4u*)&lA[cur][elt] = *(const v4u*)(gA + (size_t)rw * K + ik * 32 + cl);
        *(v4u*)&lB[cur][elt] = *(const v4u*)(gW + (size_t)rw * K + ik * 32 + cl);
      }
    }
    __syncthreads();
    v16bf af[4], bfr[2];
    for (int i = 0; i < 4; ++i) af[i]  = load_fragA(&lA[cur][0], 32, wm + i * 16, 0);
    for (int j = 0; j < 2; ++j) bfr[j] = load_fragB(&lB[cur][0], 32, wn + j * 16, 0);
    for (int i = 0; i < 4; ++i)
      for (int j = 0; j < 2; ++j)
        acc[i][j] = wmma_bf16(af[i], bfr[j], acc[i][j]);
    __syncthreads();
  }
#endif

  int hi = lane >> 4, kc = lane & 15;
  for (int i = 0; i < 4; ++i) {
    for (int j = 0; j < 2; ++j) {
      int col = bnBlk + wn + j * 16 + kc;
      for (int r = 0; r < 8; ++r) {
        int row = bmBlk + wm + i * 16 + hi * 8 + r;
        float v = acc[i][j][r];
        size_t off = (size_t)row * N + col;
        if (mode == 0) {
          ((float*)D)[off] = v;
        } else if (mode == 1) {
          ((unsigned short*)D)[off] = f2bf(v);
        } else if (mode == 2) {
          float g = 0.5f * v * (1.f + erff(v * 0.70710678118f));
          ((unsigned short*)D)[off] = f2bf(g);
        } else if (mode == 3) {
          ((float*)D)[off] = v + R[off];
        } else {
          float c = coef[(size_t)row * 8 + expert];
          ((float*)D)[off] = R[off] + c * v;
        }
      }
    }
  }
}

// ---------------- flash attention (causal), WMMA bf16 ----------------
// grid (B*H, T/64), block 128; wave handles 16 query rows; key blocks of 32
__global__ void __launch_bounds__(128, 1)
attn_kernel(const unsigned short* __restrict__ qkv, unsigned short* __restrict__ y,
            int B, int T, int H) {
  const int HD = 64;
  const int C = H * HD;     // 1024
  const int C3 = 3 * C;     // 3072
  int lane = threadIdx.x & 31;
  int wid = threadIdx.x >> 5;
  int bh = blockIdx.x;
  int b = bh / H, h = bh % H;
  int t0 = blockIdx.y * 64 + wid * 16;

  const unsigned short* qb = qkv + (size_t)(b * T) * C3 + h * HD;
  const unsigned short* kb = qb + C;
  const unsigned short* vb = qb + 2 * C;

  __shared__ __align__(16) unsigned short pl[4][16][32];

  v16bf qa[2];
  qa[0] = load_fragA(qb, C3, t0, 0);
  qa[1] = load_fragA(qb, C3, t0, 32);

  v8f o[4];
  float mrun[8], lrun[8];
  for (int r = 0; r < 8; ++r) { mrun[r] = -1e30f; lrun[r] = 0.f; }
  for (int nd = 0; nd < 4; ++nd)
    for (int r = 0; r < 8; ++r) o[nd][r] = 0.f;

  int hi = lane >> 4, kc = lane & 15;

  for (int s0 = 0; s0 < t0 + 16; s0 += 32) {
    v8f sa, sb;
    for (int r = 0; r < 8; ++r) { sa[r] = 0.f; sb[r] = 0.f; }
    for (int d0 = 0; d0 < 64; d0 += 32) {
      v16bf k0 = load_fragB(kb, C3, s0, d0);
      v16bf k1 = load_fragB(kb, C3, s0 + 16, d0);
      sa = wmma_bf16(qa[d0 >> 5], k0, sa);
      sb = wmma_bf16(qa[d0 >> 5], k1, sb);
    }
    for (int r = 0; r < 8; ++r) {
      int tq = t0 + hi * 8 + r;
      float a0 = (s0 + kc      <= tq) ? sa[r] * 0.125f : -1e30f;
      float a1 = (s0 + 16 + kc <= tq) ? sb[r] * 0.125f : -1e30f;
      float mx = fmaxf(a0, a1);
      for (int m = 8; m >= 1; m >>= 1) mx = fmaxf(mx, __shfl_xor(mx, m, 32));
      float mnew = fmaxf(mrun[r], mx);
      float corr = __expf(mrun[r] - mnew);
      float p0 = __expf(a0 - mnew);
      float p1 = __expf(a1 - mnew);
      float rs = p0 + p1;
      for (int m = 8; m >= 1; m >>= 1) rs += __shfl_xor(rs, m, 32);
      lrun[r] = lrun[r] * corr + rs;
      mrun[r] = mnew;
      for (int nd = 0; nd < 4; ++nd) o[nd][r] = o[nd][r] * corr;
      pl[wid][hi * 8 + r][kc]      = f2bf(p0);
      pl[wid][hi * 8 + r][kc + 16] = f2bf(p1);
    }
    // intra-wave cross-lane LDS dependency: wait for DS stores before reload
    asm volatile("s_wait_dscnt 0" ::: "memory");
    FragU pf;
    const unsigned short* pr = &pl[wid][lane & 15][(lane >> 4) << 3];
    pf.u[0] = *(const v4u*)pr;
    pf.u[1] = *(const v4u*)(pr + 16);
    for (int nd = 0; nd < 4; ++nd) {
      FragU vf;
      int s_base = s0 + ((lane >> 4) << 3);
      int d = nd * 16 + kc;
      for (int j = 0; j < 16; ++j)
        vf.s[j] = vb[(size_t)(s_base + j + ((j >> 3) << 3)) * C3 + d];
      o[nd] = wmma_bf16(pf.v, vf.v, o[nd]);
    }
  }

  for (int nd = 0; nd < 4; ++nd)
    for (int r = 0; r < 8; ++r) {
      int t = t0 + hi * 8 + r;
      float val = o[nd][r] / lrun[r];
      y[(size_t)(b * T + t) * C + h * HD + nd * 16 + kc] = f2bf(val);
    }
}

// ---------------- LM head: out[b,v] = dot(xf[b], wte[v]) (bandwidth-bound, f32) ----------------
__global__ void lmhead_kernel(const float* __restrict__ xf, const float* __restrict__ wte,
                              float* __restrict__ out, int V, int C) {
  int lane = threadIdx.x & 31;
  int wid = threadIdx.x >> 5;
  int v = blockIdx.x * (blockDim.x >> 5) + wid;
  if (v >= V) return;
  const float* w = wte + (size_t)v * C;
  float a0 = 0.f, a1 = 0.f;
  for (int k = lane; k < C; k += 32) {
    float ww = w[k];
    a0 += ww * xf[k];
    a1 += ww * xf[C + k];
  }
  for (int m = 16; m >= 1; m >>= 1) { a0 += __shfl_xor(a0, m, 32); a1 += __shfl_xor(a1, m, 32); }
  if (lane == 0) { out[v] = a0; out[V + v] = a1; }
}

// ---------------- host ----------------

extern "C" void kernel_launch(void* const* d_in, const int* in_sizes, int n_in,
                              void* d_out, int out_size, void* d_ws, size_t ws_size,
                              hipStream_t stream) {
  const int B = 2, T = 2048, C = 1024, H = 16, V = 32000, E = 8, FF = 4096;
  const int NT = B * T;  // 4096 tokens

  const int*   idx      = (const int*)d_in[0];
  const float* wte      = (const float*)d_in[1];
  const float* wpe      = (const float*)d_in[2];
  const float* ln1_w    = (const float*)d_in[3];
  const float* c_attn_w = (const float*)d_in[4];
  const float* c_proj_w = (const float*)d_in[5];
  const float* ln2_w    = (const float*)d_in[6];
  const float* gate_w   = (const float*)d_in[7];
  const float* W1       = (const float*)d_in[8];
  const float* W2       = (const float*)d_in[9];
  const float* lnf_w    = (const float*)d_in[10];

  char* ws = (char*)d_ws;
  size_t off = 0;
  auto alloc = [&](size_t bytes) -> void* {
    void* p = ws + off;
    off = (off + bytes + 255) & ~(size_t)255;
    return p;
  };
  float*          x    = (float*)alloc((size_t)NT * C * 4);
  unsigned short* xln1 = (unsigned short*)alloc((size_t)NT * C * 2);
  unsigned short* wqkv = (unsigned short*)alloc((size_t)3 * C * C * 2);
  unsigned short* qkv  = (unsigned short*)alloc((size_t)NT * 3 * C * 2);
  unsigned short* yatt = (unsigned short*)alloc((size_t)NT * C * 2);
  unsigned short* wprj = (unsigned short*)alloc((size_t)C * C * 2);
  float*          x2   = (float*)alloc((size_t)NT * C * 4);
  unsigned short* xln2 = (unsigned short*)alloc((size_t)NT * C * 2);
  float*          coef = (float*)alloc((size_t)NT * E * 4);
  unsigned short* w1b  = (unsigned short*)alloc((size_t)E * FF * C * 2);
  unsigned short* w2b  = (unsigned short*)alloc((size_t)E * C * FF * 2);
  unsigned short* hbuf = (unsigned short*)alloc((size_t)NT * FF * 2);
  float*          x3   = (float*)alloc((size_t)NT * C * 4);
  float*          xf   = (float*)alloc((size_t)B * C * 4);

  // stage bf16 weights
  cvt_kernel<<<1024, 256, 0, stream>>>(c_attn_w, wqkv, (long)3 * C * C);
  cvt_kernel<<<1024, 256, 0, stream>>>(c_proj_w, wprj, (long)C * C);
  cvt_kernel<<<4096, 256, 0, stream>>>(W1, w1b, (long)E * FF * C);
  cvt_kernel<<<4096, 256, 0, stream>>>(W2, w2b, (long)E * C * FF);

  embed_kernel<<<NT, 256, 0, stream>>>(idx, wte, wpe, x, T, C);
  ln_kernel<<<NT / 8, 256, 0, stream>>>(x, ln1_w, xln1, NT, 0, 1, 1);

  // qkv = ln1(x) @ c_attn_w^T  -> bf16
  gemm_kernel<<<dim3(3 * C / 128, NT / 128), 256, 0, stream>>>(
      xln1, wqkv, NT, 3 * C, C, qkv, nullptr, nullptr, 0, 1);

  attn_kernel<<<dim3(B * H, T / 64), 128, 0, stream>>>(qkv, yatt, B, T, H);

  // x2 = x + y @ c_proj_w^T
  gemm_kernel<<<dim3(C / 128, NT / 128), 256, 0, stream>>>(
      yatt, wprj, NT, C, C, x2, x, nullptr, 0, 3);

  ln_kernel<<<NT / 8, 256, 0, stream>>>(x2, ln2_w, xln2, NT, 0, 1, 1);
  router_kernel<<<NT / 8, 256, 0, stream>>>(xln2, gate_w, coef, NT);

  // MoE: x3 = x2 + sum_e coef_e * gelu(ln2 @ W1_e^T) @ W2_e^T
  for (int e = 0; e < E; ++e) {
    gemm_kernel<<<dim3(FF / 128, NT / 128), 256, 0, stream>>>(
        xln2, w1b + (size_t)e * FF * C, NT, FF, C, hbuf, nullptr, nullptr, 0, 2);
    gemm_kernel<<<dim3(C / 128, NT / 128), 256, 0, stream>>>(
        hbuf, w2b + (size_t)e * C * FF, NT, C, FF, x3, (e == 0) ? x2 : x3, coef, e, 4);
  }

  // final LN on last token of each batch only
  ln_kernel<<<1, 256, 0, stream>>>(x3, lnf_w, xf, B, T - 1, T, 0);

  // logits = lnf(x3)[last] @ wte^T
  lmhead_kernel<<<V / 8, 256, 0, stream>>>(xf, wte, (float*)d_out, V, C);
}
